// ProsodyExtractor_63531156243094
// MI455X (gfx1250) — matrix operand order repaired and checked by reading, not compiled, not run on hardware
//
#include <hip/hip_runtime.h>
#include <stdint.h>

#define NSEG 16384
#define DM   80
#define GG   192   // 3H gate width
#define HH   64
#define K2IH 48    // K padded 80->96, in bf16 pairs
#define K2HH 32    // K=64, in bf16 pairs

typedef __attribute__((ext_vector_type(16))) __bf16 v16bf;
typedef __attribute__((ext_vector_type(8)))  float  v8f;

#ifndef __has_builtin
#define __has_builtin(x) 0
#endif

// f32 -> packed bf16 pair. Cast through __bf16 so the backend selects the
// native v_cvt(_pk)_bf16_f32 when available; expands to bit ops otherwise.
static __device__ __forceinline__ uint32_t pkbf(float lo, float hi) {
  union { __bf16 h[2]; uint32_t u; } c;
  c.h[0] = (__bf16)lo;
  c.h[1] = (__bf16)hi;
  return c.u;
}

static __device__ __forceinline__ v16bf as16(const uint32_t* u) {
  union { uint32_t x[8]; v16bf v; } c;
#pragma unroll
  for (int i = 0; i < 8; ++i) c.x[i] = u[i];
  return c.v;
}

static __device__ __forceinline__ v8f wmma_bf16(const uint32_t* a, const uint32_t* b, v8f c) {
  return __builtin_amdgcn_wmma_f32_16x16x32_bf16(false, as16(a), false, as16(b),
                                                 (short)0, c, false, false);
}

static __device__ __forceinline__ v8f splat8(float x) {
  v8f r;
#pragma unroll
  for (int i = 0; i < 8; ++i) r[i] = x;
  return r;
}

static __device__ __forceinline__ float sigm(float x) {
  return __builtin_amdgcn_rcpf(1.f + __expf(-x));
}
static __device__ __forceinline__ float tanh_(float x) {
#if __has_builtin(__builtin_amdgcn_tanhf)
  return __builtin_amdgcn_tanhf(x);
#else
  x = fminf(fmaxf(x, -15.f), 15.f);
  float e = __expf(2.f * x);
  return (e - 1.f) * __builtin_amdgcn_rcpf(e + 1.f);
#endif
}

// ---------------- setup kernels ----------------

__global__ void k_init(int* hist) { if (threadIdx.x < 32) hist[threadIdx.x] = 0; }

// per-utterance exclusive prefix of durations + global length histogram
__global__ void k_scan(const int* __restrict__ dur, int* __restrict__ starts,
                       int* __restrict__ hist) {
  __shared__ int sh[256];
  int b = blockIdx.x, i = threadIdx.x;
  int d = dur[b * 256 + i];
  sh[i] = d; __syncthreads();
  for (int off = 1; off < 256; off <<= 1) {
    int tv = (i >= off) ? sh[i - off] : 0;
    __syncthreads(); sh[i] += tv; __syncthreads();
  }
  starts[b * 256 + i] = sh[i] - d;
  atomicAdd(&hist[d & 31], 1);
}

// keep[i] = 1 iff segment i's length equals its sort-group (of 64) max.
// Element with value v at stable rank r (descending) is kept iff
// floor(r/64)*64 >= cnt_gt(v)  (group leader lies inside the run of value v).
__global__ void k_keep(const int* __restrict__ dur, const int* __restrict__ hist,
                       float* __restrict__ keep) {
  __shared__ int sh[256];
  __shared__ int sbase;
  int v = blockIdx.x, tid = threadIdx.x;
  int cgt = 0;
  for (int u = v + 1; u < 32; ++u) cgt += hist[u];
  if (tid == 0) sbase = 0;
  __syncthreads();
  for (int c0 = 0; c0 < NSEG; c0 += 256) {
    int i = c0 + tid;
    int f = (dur[i] == v) ? 1 : 0;
    sh[tid] = f; __syncthreads();
    for (int off = 1; off < 256; off <<= 1) {
      int tv = (tid >= off) ? sh[tid - off] : 0;
      __syncthreads(); sh[tid] += tv; __syncthreads();
    }
    int incl = sh[tid];
    int tot = sh[255];
    int base = sbase;
    if (f) {
      int rank = cgt + base + incl - 1;
      keep[i] = (((rank / 64) * 64) >= cgt) ? 1.f : 0.f;
    }
    __syncthreads();
    if (tid == 0) sbase += tot;
    __syncthreads();
  }
}

// ---------------- main GRU kernel ----------------
// 8 waves / block, each wave owns a tile of 16 segments (M dim of WMMA).
// Gates g = 192 channels -> 12 N-tiles: 0..7 = r,z (x & h fused), 8..11 = n (xn / hn apart).

__global__ __launch_bounds__(256) void gru_kernel(
    const float* __restrict__ mels, const int* __restrict__ dur,
    const float* cw1, const float* cb1, const float* g1, const float* be1,
    const float* cw2, const float* cb2, const float* g2, const float* be2,
    const float* __restrict__ Wihf, const float* __restrict__ Whhf,
    const float* __restrict__ bihf, const float* __restrict__ bhhf,
    const float* __restrict__ Wihb, const float* __restrict__ bihb,
    const float* __restrict__ bhhb, const int* __restrict__ starts,
    const float* __restrict__ keep, float* __restrict__ out, int T)
{
  __shared__ uint32_t sWih[GG * K2IH];   // fwd input weights, bf16 pairs (B layout friendly)
  __shared__ uint32_t sWib[GG * K2IH];   // bwd input weights
  __shared__ uint32_t sWhh[GG * K2HH];   // fwd recurrent weights
  __shared__ float sBsum[128], sBxn[64], sBhn[64];
  __shared__ float sBsumB[128], sBxnB[64], sBhnB[64];
  __shared__ float sH[8][16 * 64];       // per-wave h transpose buffer [M][ch]
  __shared__ int   sLen[8][16];
  __shared__ float sKeep[8][16];
  __shared__ int   sMaxLen;

  const int tid = threadIdx.x;

  // Pack weights into LDS: uint = bf16 pair (W[g][k], W[g][k+1]); K zero-padded to 96.
  for (int idx = tid; idx < GG * K2IH; idx += 256) {
    int g = idx / K2IH, k = (idx % K2IH) * 2;
    float a = (k < DM) ? Wihf[g * DM + k] : 0.f;
    float b = (k + 1 < DM) ? Wihf[g * DM + k + 1] : 0.f;
    sWih[idx] = pkbf(a, b);
    a = (k < DM) ? Wihb[g * DM + k] : 0.f;
    b = (k + 1 < DM) ? Wihb[g * DM + k + 1] : 0.f;
    sWib[idx] = pkbf(a, b);
  }
  for (int idx = tid; idx < GG * K2HH; idx += 256) {
    int g = idx / K2HH, k = (idx % K2HH) * 2;
    sWhh[idx] = pkbf(Whhf[g * HH + k], Whhf[g * HH + k + 1]);
  }
  for (int i = tid; i < 128; i += 256) {
    sBsum[i]  = bihf[i] + bhhf[i];      // r,z: x-bias + h-bias fused
    sBsumB[i] = bihb[i] + bhhb[i];
  }
  for (int i = tid; i < 64; i += 256) {
    sBxn[i]  = bihf[128 + i]; sBhn[i]  = bhhf[128 + i];
    sBxnB[i] = bihb[128 + i]; sBhnB[i] = bhhb[128 + i];
  }
  if (tid == 0) sMaxLen = 0;
  __syncthreads();

  const int wave = tid >> 5, lane = tid & 31, lanN = lane & 15;
  const int hi = (lane & 16) ? 1 : 0;
  const int tile = blockIdx.x * 8 + wave;
  const int segbase = tile * 16;
  const int mySeg = segbase + lanN;
  const int myB = mySeg >> 8;
  const int myStart = starts[mySeg];
  const int myLen = dur[mySeg];
  if (!hi) { sLen[wave][lanN] = myLen; sKeep[wave][lanN] = keep[mySeg]; }
  atomicMax(&sMaxLen, myLen);
  __syncthreads();

  int lenv[8]; float keepv[8];
#pragma unroll
  for (int v = 0; v < 8; ++v) {
    int M = hi ? v + 8 : v;
    lenv[v] = sLen[wave][M];
    keepv[v] = sKeep[wave][M];
  }

  // conv+BN(eval)+ReLU collapses to two scalar affines
  const float inv = 1.0f / sqrtf(1.0f + 1e-5f);
  const float C1 = cw1[0] * g1[0] * inv, D1 = cb1[0] * g1[0] * inv + be1[0];
  const float C2 = cw2[0] * g2[0] * inv, D2 = cb2[0] * g2[0] * inv + be2[0];

  v8f hA[4];                              // h state, accumulator layout (M,N)=(seg,ch)
#pragma unroll
  for (int tj = 0; tj < 4; ++tj) hA[tj] = splat8(0.f);

  const int LMAX = sMaxLen;
  float* hrow = sH[wave];

  for (int t = 0; t < LMAX; ++t) {
    // publish h (accumulator layout -> [M][ch]) for the A-layout transpose
#pragma unroll
    for (int tj = 0; tj < 4; ++tj)
#pragma unroll
      for (int v = 0; v < 8; ++v) {
        int M = hi ? v + 8 : v;
        hrow[M * 64 + tj * 16 + lanN] = hA[tj][v];
      }
    __syncthreads();

    // A operand: 16 frames, K padded to 96 (3 chunks of 32). ISA 16-bit A layout:
    // lane<16 holds K {0..7,16..23}+32*kc of row M=lane; lane>=16 holds K {8..15,24..31}.
    uint32_t AX[3][8];
    const bool tv = (t < myLen);
    const float* row = mels + ((size_t)myB * T + (size_t)(myStart + t)) * DM;
#pragma unroll
    for (int kc = 0; kc < 3; ++kc)
#pragma unroll
      for (int j = 0; j < 8; ++j) {
        int k = kc * 32 + ((j < 4) ? 0 : 16) + (hi ? 8 : 0) + 2 * (j & 3);
        float f0 = 0.f, f1 = 0.f;
        if (tv) {
          if (k < DM)     { float x = row[k];     float y = fmaxf(fmaf(C1,x,D1),0.f); f0 = fmaxf(fmaf(C2,y,D2),0.f); }
          if (k + 1 < DM) { float x = row[k + 1]; float y = fmaxf(fmaf(C1,x,D1),0.f); f1 = fmaxf(fmaf(C2,y,D2),0.f); }
        }
        AX[kc][j] = pkbf(f0, f1);
      }
    // A operand for h (K=64 -> 2 chunks)
    uint32_t AH[2][8];
#pragma unroll
    for (int kc = 0; kc < 2; ++kc)
#pragma unroll
      for (int j = 0; j < 8; ++j) {
        int k2 = kc * 16 + (j & 3) + ((j < 4) ? 0 : 8) + (hi ? 4 : 0);
        AH[kc][j] = pkbf(hrow[lanN * 64 + 2 * k2], hrow[lanN * 64 + 2 * k2 + 1]);
      }

    v8f aRZ[8], aX[4], aHc[4];
#pragma unroll
    for (int nt = 0; nt < 8; ++nt) aRZ[nt] = splat8(sBsum[nt * 16 + lanN]);
#pragma unroll
    for (int n2 = 0; n2 < 4; ++n2) { aX[n2] = splat8(sBxn[n2 * 16 + lanN]);
                                     aHc[n2] = splat8(sBhn[n2 * 16 + lanN]); }

    uint32_t Bv[8];
#pragma unroll
    for (int kc = 0; kc < 3; ++kc) {
#pragma unroll
      for (int nt = 0; nt < 12; ++nt) {
        const uint32_t* p = sWih + (nt * 16 + lanN) * K2IH + kc * 16 + (hi ? 4 : 0);
#pragma unroll
        for (int v = 0; v < 8; ++v) Bv[v] = p[(v < 4) ? v : v + 4];
        if (nt < 8) aRZ[nt]     = wmma_bf16(AX[kc], Bv, aRZ[nt]);
        else        aX[nt - 8]  = wmma_bf16(AX[kc], Bv, aX[nt - 8]);
      }
    }
#pragma unroll
    for (int kc = 0; kc < 2; ++kc) {
#pragma unroll
      for (int nt = 0; nt < 12; ++nt) {
        const uint32_t* p = sWhh + (nt * 16 + lanN) * K2HH + kc * 16 + (hi ? 4 : 0);
#pragma unroll
        for (int v = 0; v < 8; ++v) Bv[v] = p[(v < 4) ? v : v + 4];
        if (nt < 8) aRZ[nt]     = wmma_bf16(AH[kc], Bv, aRZ[nt]);
        else        aHc[nt - 8] = wmma_bf16(AH[kc], Bv, aHc[nt - 8]);
      }
    }

    // gates + masked state update (scan: where(valid, h_new, h))
#pragma unroll
    for (int tj = 0; tj < 4; ++tj)
#pragma unroll
      for (int v = 0; v < 8; ++v) {
        float r = sigm(aRZ[tj][v]);
        float z = sigm(aRZ[4 + tj][v]);
        float n = tanh_(aX[tj][v] + r * aHc[tj][v]);
        float ho = hA[tj][v];
        float hn = fmaf(z, ho - n, n);          // (1-z)*n + z*ho
        hA[tj][v] = (t < lenv[v]) ? hn : ho;
      }
    __syncthreads();
  }

  // backward direction: one GRU step from h0=0 on the last valid frame (Whh_b unused)
  {
    uint32_t AX[3][8];
    const float* row = mels + ((size_t)myB * T + (size_t)(myStart + myLen - 1)) * DM;
#pragma unroll
    for (int kc = 0; kc < 3; ++kc)
#pragma unroll
      for (int j = 0; j < 8; ++j) {
        int k = kc * 32 + ((j < 4) ? 0 : 16) + (hi ? 8 : 0) + 2 * (j & 3);
        float f0 = 0.f, f1 = 0.f;
        if (k < DM)     { float x = row[k];     float y = fmaxf(fmaf(C1,x,D1),0.f); f0 = fmaxf(fmaf(C2,y,D2),0.f); }
        if (k + 1 < DM) { float x = row[k + 1]; float y = fmaxf(fmaf(C1,x,D1),0.f); f1 = fmaxf(fmaf(C2,y,D2),0.f); }
        AX[kc][j] = pkbf(f0, f1);
      }
    v8f aRZ[8], aX[4];
#pragma unroll
    for (int nt = 0; nt < 8; ++nt) aRZ[nt] = splat8(sBsumB[nt * 16 + lanN]);
#pragma unroll
    for (int n2 = 0; n2 < 4; ++n2) aX[n2] = splat8(sBxnB[n2 * 16 + lanN]);
    uint32_t Bv[8];
#pragma unroll
    for (int kc = 0; kc < 3; ++kc)
#pragma unroll
      for (int nt = 0; nt < 12; ++nt) {
        const uint32_t* p = sWib + (nt * 16 + lanN) * K2IH + kc * 16 + (hi ? 4 : 0);
#pragma unroll
        for (int v = 0; v < 8; ++v) Bv[v] = p[(v < 4) ? v : v + 4];
        if (nt < 8) aRZ[nt]    = wmma_bf16(AX[kc], Bv, aRZ[nt]);
        else        aX[nt - 8] = wmma_bf16(AX[kc], Bv, aX[nt - 8]);
      }

#pragma unroll
    for (int tj = 0; tj < 4; ++tj) {
      float hn0 = sBhnB[tj * 16 + lanN];   // hg_n = bhh_b (h0 == 0)
#pragma unroll
      for (int v = 0; v < 8; ++v) {
        float r = sigm(aRZ[tj][v]);
        float z = sigm(aRZ[4 + tj][v]);
        float n = tanh_(aX[tj][v] + r * hn0);
        float hb = n - z * n;               // (1-z)*n
        int M = hi ? v + 8 : v;
        size_t o = (size_t)(segbase + M) * 128 + tj * 16 + lanN;
        out[o]      = hA[tj][v] * keepv[v];
        out[o + 64] = hb * keepv[v];
      }
    }
  }
}

// ---------------- host entry ----------------

extern "C" void kernel_launch(void* const* d_in, const int* in_sizes, int n_in,
                              void* d_out, int out_size, void* d_ws, size_t ws_size,
                              hipStream_t stream) {
  (void)n_in; (void)out_size; (void)ws_size;
  const float* mels = (const float*)d_in[0];
  const int*   dur  = (const int*)d_in[1];
  const float* cw1 = (const float*)d_in[2];
  const float* cb1 = (const float*)d_in[3];
  const float* g1  = (const float*)d_in[4];
  const float* be1 = (const float*)d_in[5];
  const float* cw2 = (const float*)d_in[6];
  const float* cb2 = (const float*)d_in[7];
  const float* g2  = (const float*)d_in[8];
  const float* be2 = (const float*)d_in[9];
  const float* Wihf = (const float*)d_in[10];
  const float* Whhf = (const float*)d_in[11];
  const float* bihf = (const float*)d_in[12];
  const float* bhhf = (const float*)d_in[13];
  const float* Wihb = (const float*)d_in[14];
  // d_in[15] = Whh_b: mathematically unused (backward step starts from h0 == 0)
  const float* bihb = (const float*)d_in[16];
  const float* bhhb = (const float*)d_in[17];
  float* out = (float*)d_out;

  int T = in_sizes[0] / (64 * DM);

  int*   hist   = (int*)d_ws;
  int*   starts = hist + 32;
  float* keepA  = (float*)(starts + NSEG);

  k_init<<<1, 32, 0, stream>>>(hist);
  k_scan<<<64, 256, 0, stream>>>(dur, starts, hist);
  k_keep<<<32, 256, 0, stream>>>(dur, hist, keepA);
  gru_kernel<<<128, 256, 0, stream>>>(mels, dur, cw1, cb1, g1, be1, cw2, cb2, g2, be2,
                                      Wihf, Whhf, bihf, bhhf, Wihb, bihb, bhhb,
                                      starts, keepA, out, T);
}
